// Linear_48077863911552
// MI455X (gfx1250) — matrix operand order repaired and checked by reading, not compiled, not run on hardware
//
#include <hip/hip_runtime.h>

typedef __attribute__((ext_vector_type(16))) _Float16 v16h;
typedef __attribute__((ext_vector_type(8)))  float    v8f;

#define D1 1152   // 128*(1+3+5)

static __device__ __forceinline__ v16h hsplat(_Float16 s) {
  v16h v;
#pragma unroll
  for (int i = 0; i < 16; ++i) v[i] = s;
  return v;
}

static __device__ __forceinline__ v8f wmma_f16(v16h a, v16h b, v8f c) {
  return __builtin_amdgcn_wmma_f32_16x16x32_f16(
      false, a, false, b, (short)0, c, false, false);
}

static __device__ __forceinline__ unsigned int pk2(float a, float b) {
  union { _Float16 h[2]; unsigned int u; } t;
  t.h[0] = (_Float16)a;
  t.h[1] = (_Float16)b;
  return t.u;
}

// LDS dword offset of the (even) position pair holding K=k of plane p, row r.
// Fragment order: lanes0-15 hold K {0-15}, lanes16-31 {16-31}.
static __device__ __forceinline__ int lds_dw(int p, int row, int k) {
  int c2  = k & 31;
  int kk  = k >> 5;
  int lg  = c2 >> 4;
  int pos = c2 & 15;
  return ((p * 128 + kk * 32 + lg * 16 + row) << 3) + (pos >> 1);
}

// ---------------------------------------------------------------------------
// Pre-pass: w (f32, 8 x [u=128][w=128]) -> f16 transposed WT_k[n][u].
// ---------------------------------------------------------------------------
__global__ __launch_bounds__(256) void convert_w_kernel(
    const float* __restrict__ w, _Float16* __restrict__ wt) {
  int t = blockIdx.x * 256 + threadIdx.x;
  float v = w[t];
  int k = t >> 14;
  int r = t & 16383;
  int u = r >> 7;
  int n = r & 127;
  wt[(k << 14) + (n << 7) + u] = (_Float16)v;
}

// ---------------------------------------------------------------------------
// Main tensor-product kernel: 1 block = 16 batch rows, 8 waves, wave w owns
// output channels [16w,16w+16). 16 WMMAs per K-step x 4 K-steps = 64 WMMAs.
// K-step split into two phases (f0/f1-driven, then f2-driven) to cap A-frag
// liveness at ~40 VGPRs and stay under the 256-VGPR boundary.
// ---------------------------------------------------------------------------
__global__ __launch_bounds__(256) void tp_kernel(
    const float* __restrict__ x1, const float* __restrict__ x2,
    const _Float16* __restrict__ wt, float* __restrict__ out) {
  __shared__ v16h  ldsAv[9 * 128];   // 9 planes x (4 kk x 2 lg x 16 rows) frags
  __shared__ float ldsX2[16 * 9];
  unsigned int* ldsU = (unsigned int*)ldsAv;

  const int tid   = threadIdx.x;
  const int zbase = blockIdx.x * 16;

  if (tid < 144) ldsX2[tid] = x2[zbase * 9 + tid];

  // ---- Staging: 18 quads/thread (4 consecutive K of one plane,row each) ----
#pragma unroll
  for (int j = 0; j < 2; ++j) {       // plane 0, stride 1: coalesced float4
    int qi  = tid + j * 256;
    int row = qi >> 5;
    int k   = (qi & 31) << 2;
    const float* g = x1 + (size_t)(zbase + row) * D1 + k;
    float4 v = *(const float4*)g;
    uint2 q;
    q.x = pk2(v.x, v.y);
    q.y = pk2(v.z, v.w);
    *(uint2*)(ldsU + lds_dw(0, row, k)) = q;
  }
#pragma unroll
  for (int j = 2; j < 8; ++j) {       // planes 1..3 (x1_1, stride 3)
    int qi  = tid + j * 256;
    int p   = qi >> 9;
    int r2  = qi & 511;
    int row = r2 >> 5;
    int k   = (r2 & 31) << 2;
    const float* g = x1 + (size_t)(zbase + row) * D1 + 128 + k * 3 + (p - 1);
    uint2 q;
    q.x = pk2(g[0], g[3]);
    q.y = pk2(g[6], g[9]);
    *(uint2*)(ldsU + lds_dw(p, row, k)) = q;
  }
#pragma unroll
  for (int j = 8; j < 18; ++j) {      // planes 4..8 (x1_2, stride 5)
    int qi  = tid + j * 256;
    int p   = qi >> 9;
    int r2  = qi & 511;
    int row = r2 >> 5;
    int k   = (r2 & 31) << 2;
    const float* g = x1 + (size_t)(zbase + row) * D1 + 512 + k * 5 + (p - 4);
    uint2 q;
    q.x = pk2(g[0], g[5]);
    q.y = pk2(g[10], g[15]);
    *(uint2*)(ldsU + lds_dw(p, row, k)) = q;
  }
  __syncthreads();

  const int lane = tid & 31;
  const int wave = tid >> 5;
  const int m    = lane & 15;
  const int hi   = lane >> 4;

  // Row scalars only needed (as f16) for the y4/y7 pre-GEMM contractions.
  const float is3 = 0.57735026919f, is5 = 0.44721359550f;
  _Float16 hy1[3], hy2[5];
#pragma unroll
  for (int i = 0; i < 3; ++i) hy1[i] = (_Float16)(ldsX2[m * 9 + 1 + i] * is3);
#pragma unroll
  for (int i = 0; i < 5; ++i) hy2[i] = (_Float16)(ldsX2[m * 9 + 4 + i] * is5);

  const int ncol = wave * 16 + m;
  const _Float16* wb = wt + (ncol << 7) + (hi << 4);

  v8f accY = {};                                    // W4*y4 + W7*y7
  v8f t0 = {}, t1 = {}, t2 = {};                    // W0/W1/W2 ^T x1_0
  v8f t3_0 = {}, t3_1 = {}, t3_2 = {};              // W3^T x1_1_j
  v8f t5_0 = {}, t5_1 = {}, t5_2 = {};              // W5^T x1_1_j (cross via D)
  v8f t6_0 = {}, t6_1 = {}, t6_2 = {}, t6_3 = {}, t6_4 = {};  // W6^T x1_2_j

#pragma unroll
  for (int kk = 0; kk < 4; ++kk) {
    const int ab = kk * 32 + hi * 16 + m;
    const int ko = kk * 32;

    // ---------- Phase 1: planes 0..3 (f0, f1a..c) ----------
    {
      v16h f0  = ldsAv[0 * 128 + ab];
      v16h f1a = ldsAv[1 * 128 + ab];
      v16h f1b = ldsAv[2 * 128 + ab];
      v16h f1c = ldsAv[3 * 128 + ab];

      v16h b0 = *(const v16h*)(wb + 0 * 16384 + ko);
      t0 = wmma_f16(f0, b0, t0);
      v16h b1 = *(const v16h*)(wb + 1 * 16384 + ko);
      t1 = wmma_f16(f0, b1, t1);
      v16h b2 = *(const v16h*)(wb + 2 * 16384 + ko);
      t2 = wmma_f16(f0, b2, t2);

      v16h b4 = *(const v16h*)(wb + 4 * 16384 + ko);
      v16h y4 = f1a * hsplat(hy1[0]) + f1b * hsplat(hy1[1]) + f1c * hsplat(hy1[2]);
      accY = wmma_f16(y4, b4, accY);

      v16h b3 = *(const v16h*)(wb + 3 * 16384 + ko);
      t3_0 = wmma_f16(f1a, b3, t3_0);
      t3_1 = wmma_f16(f1b, b3, t3_1);
      t3_2 = wmma_f16(f1c, b3, t3_2);
      v16h b5 = *(const v16h*)(wb + 5 * 16384 + ko);
      t5_0 = wmma_f16(f1a, b5, t5_0);
      t5_1 = wmma_f16(f1b, b5, t5_1);
      t5_2 = wmma_f16(f1c, b5, t5_2);
    }
    // ---------- Phase 2: planes 4..8 (f2a..e) ----------
    {
      v16h f2a = ldsAv[4 * 128 + ab];
      v16h f2b = ldsAv[5 * 128 + ab];
      v16h f2c = ldsAv[6 * 128 + ab];
      v16h f2d = ldsAv[7 * 128 + ab];
      v16h f2e = ldsAv[8 * 128 + ab];

      v16h b7 = *(const v16h*)(wb + 7 * 16384 + ko);
      v16h y7 = f2a * hsplat(hy2[0]) + f2b * hsplat(hy2[1]) + f2c * hsplat(hy2[2])
              + f2d * hsplat(hy2[3]) + f2e * hsplat(hy2[4]);
      accY = wmma_f16(y7, b7, accY);

      v16h b6 = *(const v16h*)(wb + 6 * 16384 + ko);
      t6_0 = wmma_f16(f2a, b6, t6_0);
      t6_1 = wmma_f16(f2b, b6, t6_1);
      t6_2 = wmma_f16(f2c, b6, t6_2);
      t6_3 = wmma_f16(f2d, b6, t6_3);
      t6_4 = wmma_f16(f2e, b6, t6_4);
    }
  }

  // ---- Epilogue: all per-row x2 scalings on D tiles + normalize + NT stores.
  const float nA = 0.05103103630798287f;  // 1/sqrt(3*128)
  const float nB = 0.0625f;               // 1/sqrt(2*128)
  const float cw = 0.70710678119f * 0.05103103630798287f;  // nA/sqrt(2)
#pragma unroll
  for (int r = 0; r < 8; ++r) {
    int row = hi * 8 + r;                 // D-tile row for this lane half
    int z   = zbase + row;
    const float* xr = ldsX2 + row * 9;    // uniform per half-wave -> broadcast
    float s0  = xr[0];
    float s0A = s0 * nA, s0B = s0 * nB;
    float t1n = t1[r] * nA, t2n = t2[r] * nB;
    float c0 = xr[1] * cw, c1 = xr[2] * cw, c2 = xr[3] * cw;
    float* o  = out + (size_t)z * D1;
    __builtin_nontemporal_store(fmaf(s0A, t0[r], accY[r] * nA), o + ncol);
    float* o1 = o + 128 + ncol * 3;
    __builtin_nontemporal_store(
        fmaf(xr[1], t1n, fmaf(s0A, t3_0[r], fmaf(c2, t5_1[r], -(c1 * t5_2[r])))), o1 + 0);
    __builtin_nontemporal_store(
        fmaf(xr[2], t1n, fmaf(s0A, t3_1[r], fmaf(c0, t5_2[r], -(c2 * t5_0[r])))), o1 + 1);
    __builtin_nontemporal_store(
        fmaf(xr[3], t1n, fmaf(s0A, t3_2[r], fmaf(c1, t5_0[r], -(c0 * t5_1[r])))), o1 + 2);
    float* o2 = o + 512 + ncol * 5;
    __builtin_nontemporal_store(fmaf(xr[4], t2n, s0B * t6_0[r]), o2 + 0);
    __builtin_nontemporal_store(fmaf(xr[5], t2n, s0B * t6_1[r]), o2 + 1);
    __builtin_nontemporal_store(fmaf(xr[6], t2n, s0B * t6_2[r]), o2 + 2);
    __builtin_nontemporal_store(fmaf(xr[7], t2n, s0B * t6_3[r]), o2 + 3);
    __builtin_nontemporal_store(fmaf(xr[8], t2n, s0B * t6_4[r]), o2 + 4);
  }
}

extern "C" void kernel_launch(void* const* d_in, const int* in_sizes, int n_in,
                              void* d_out, int out_size, void* d_ws, size_t ws_size,
                              hipStream_t stream) {
  const float* x1 = (const float*)d_in[0];
  const float* x2 = (const float*)d_in[1];
  const float* w  = (const float*)d_in[2];
  float* out = (float*)d_out;
  _Float16* wt = (_Float16*)d_ws;   // 256 KB scratch

  int Z = in_sizes[0] / D1;         // 65536

  convert_w_kernel<<<(8 * 16384) / 256, 256, 0, stream>>>(w, wt);
  tp_kernel<<<Z / 16, 256, 0, stream>>>(x1, x2, wt, out);
}
